// MaskedLSTM_63350767616194
// MI455X (gfx1250) — matrix operand order repaired and checked by reading, not compiled, not run on hardware
//
#include <hip/hip_runtime.h>
#include <hip/hip_bf16.h>
#include <math.h>

// ---- sizes from the reference ----
#define B_ 128
#define S_ 1024
#define I_ 256
#define H_ 512
#define G_ 2048   // 4*H
#define O_ 128

typedef __bf16 bf16_t;
typedef __attribute__((ext_vector_type(16))) __bf16 bf16x16;
typedef __attribute__((ext_vector_type(8)))  __bf16 bf16x8;
typedef __attribute__((ext_vector_type(8)))  float  f32x8;
typedef __attribute__((ext_vector_type(4)))  int    v4i;

// ---------------- async global -> LDS (CDNA5, ASYNCcnt-tracked) ----------------
#if defined(__has_builtin)
#if __has_builtin(__builtin_amdgcn_global_load_async_to_lds_b128)
#define ASYNC_VIA_BUILTIN 1
#endif
#if __has_builtin(__builtin_amdgcn_s_wait_asynccnt)
#define WAIT_VIA_BUILTIN 1
#endif
#endif

typedef __attribute__((address_space(1))) v4i* gv4i_p;   // global <4 x i32>*
typedef __attribute__((address_space(3))) v4i* lv4i_p;   // LDS    <4 x i32>*

__device__ __forceinline__ void copy16_async(bf16_t* l, const bf16_t* g) {
#if defined(ASYNC_VIA_BUILTIN)
  __builtin_amdgcn_global_load_async_to_lds_b128(
      (gv4i_p)(uintptr_t)g, (lv4i_p)(uintptr_t)l, 0, 0);
#else
  unsigned lds_addr = (unsigned)(uintptr_t)l;
  asm volatile("global_load_async_to_lds_b128 %0, %1, off"
               :: "v"(lds_addr), "v"(g) : "memory");
#endif
}
__device__ __forceinline__ void wait_async_all() {
#if defined(WAIT_VIA_BUILTIN)
  __builtin_amdgcn_s_wait_asynccnt(0);
#else
  asm volatile("s_wait_asynccnt 0x0" ::: "memory");
#endif
}

// ---------------- device-scope grid barrier (monotonic counter) ----------------
__device__ __forceinline__ void grid_barrier(unsigned* bar, unsigned target, int tid) {
  __syncthreads();
  if (tid == 0) {
    __hip_atomic_fetch_add(bar, 1u, __ATOMIC_RELEASE, __HIP_MEMORY_SCOPE_AGENT);
    while (__hip_atomic_load(bar, __ATOMIC_ACQUIRE, __HIP_MEMORY_SCOPE_AGENT) < target) {
      __builtin_amdgcn_s_sleep(2);
    }
  }
  __syncthreads();
}

// ---------------- WMMA fragment loads (ISA 7.12.2 layouts) ----------------
__device__ __forceinline__ bf16x16 frag_cat(bf16x8 lo, bf16x8 hi) {
  return __builtin_shufflevector(lo, hi, 0,1,2,3,4,5,6,7,8,9,10,11,12,13,14,15);
}
__device__ __forceinline__ bf16x16 load_a_frag(const bf16_t* sA, int m0, int lane) {
  const bf16_t* p = sA + (m0 + (lane & 15)) * 32 + (lane >> 4) * 8;
  bf16x8 lo = *(const bf16x8*)(p);
  bf16x8 hi = *(const bf16x8*)(p + 16);
  return frag_cat(lo, hi);
}
__device__ __forceinline__ bf16x16 load_b_frag(const bf16_t* sB, int n0, int lane) {
  const bf16_t* p = sB + (n0 + (lane & 15)) * 32 + (lane >> 4) * 8;
  bf16x8 lo = *(const bf16x8*)(p);
  bf16x8 hi = *(const bf16x8*)(p + 16);
  return frag_cat(lo, hi);
}

// ---------------- synchronous tile loaders (phase-1 / fc) ----------------
__device__ __forceinline__ void load_tile_f32(bf16_t* s, const float* g, long lda, int tid) {
  const int r  = tid >> 3;
  const int cc = (tid & 7) * 4;
  #pragma unroll
  for (int rr = 0; rr < 128; rr += 32) {
    const float4 v = *(const float4*)(g + (long)(r + rr) * lda + cc);
    bf16_t* sp = s + (r + rr) * 32 + cc;
    sp[0] = (bf16_t)v.x; sp[1] = (bf16_t)v.y; sp[2] = (bf16_t)v.z; sp[3] = (bf16_t)v.w;
  }
}
__device__ __forceinline__ void load_tile_bf16(bf16_t* s, const bf16_t* g, long lda, int tid) {
  const int r  = tid >> 2;
  const int cc = (tid & 3) * 8;
  #pragma unroll
  for (int rr = 0; rr < 128; rr += 64) {
    bf16x8 v = *(const bf16x8*)(g + (long)(r + rr) * lda + cc);
    *(bf16x8*)(s + (r + rr) * 32 + cc) = v;
  }
}

// ---------------- Phase 1: x_proj = bf16( x @ W_ih^T + b_ih ) ----------------
__global__ __launch_bounds__(256)
void xproj_gemm_kernel(const float* __restrict__ x, const bf16_t* __restrict__ wih,
                       const float* __restrict__ b_ih, bf16_t* __restrict__ xp)
{
  __shared__ __align__(16) bf16_t sA[128 * 32];
  __shared__ __align__(16) bf16_t sB[128 * 32];
  const int tid = threadIdx.x, lane = tid & 31, wave = tid >> 5;
  const long m0 = (long)blockIdx.x * 128;
  const int  n0 = blockIdx.y * 128;
  const int  wm = (wave & 3) * 32, wn = (wave >> 2) * 64;

  f32x8 acc[2][4] = {};
  for (int k0 = 0; k0 < I_; k0 += 32) {
    load_tile_f32(sA, x + m0 * I_ + k0, I_, tid);
    load_tile_bf16(sB, wih + (long)n0 * I_ + k0, I_, tid);
    __syncthreads();
    bf16x16 af[2], bfv[4];
    #pragma unroll
    for (int i = 0; i < 2; ++i) af[i] = load_a_frag(sA, wm + i * 16, lane);
    #pragma unroll
    for (int j = 0; j < 4; ++j) bfv[j] = load_b_frag(sB, wn + j * 16, lane);
    #pragma unroll
    for (int i = 0; i < 2; ++i)
      #pragma unroll
      for (int j = 0; j < 4; ++j)
        acc[i][j] = __builtin_amdgcn_wmma_f32_16x16x32_bf16(
            false, af[i], false, bfv[j], (short)0, acc[i][j], false, false);
    __syncthreads();
  }
  #pragma unroll
  for (int i = 0; i < 2; ++i) {
    #pragma unroll
    for (int j = 0; j < 4; ++j) {
      const int  nc = n0 + wn + j * 16 + (lane & 15);
      const float bias = b_ih[nc];
      const long mr = m0 + wm + i * 16 + (lane >> 4) * 8;
      #pragma unroll
      for (int r = 0; r < 8; ++r)
        xp[(mr + r) * G_ + nc] = (bf16_t)(acc[i][j][r] + bias);
    }
  }
}

// ---------------- Persistent recurrent kernel: all 1024 steps in one launch ----------------
// 16 co-resident blocks; block nb owns h columns [nb*32, +32) and the 4 matching
// gate groups. Per step: async double-buffered GEMM (M=128, N=128, K=512) +
// fused cell update, then a device-scope grid barrier. h ping-pongs in global bf16.
__global__ __launch_bounds__(256)
void lstm_persistent_kernel(bf16_t* __restrict__ h0g, bf16_t* __restrict__ h1g,
                            float* __restrict__ c, const bf16_t* __restrict__ whh,
                            const bf16_t* __restrict__ xp, const float* __restrict__ b_hh,
                            const int* __restrict__ mask, unsigned* __restrict__ bar)
{
  __shared__ __align__(16) float sg[128 * 128];   // 64KB gates; first 32KB aliased as tiles
  bf16_t* sAb = (bf16_t*)sg;        // [2][128*32]
  bf16_t* sBb = sAb + 2 * 4096;     // [2][128*32]
  const int tid = threadIdx.x, lane = tid & 31, wave = tid >> 5;
  const int nb = blockIdx.x;
  const int wm = (wave & 3) * 32, wn = (wave >> 2) * 64;
  const int r = tid >> 2, cc = (tid & 3) * 8;     // async-copy coords: 16B per chunk

  for (int t = 0; t < S_; ++t) {
    const bf16_t* h_in  = (t & 1) ? h1g : h0g;
    bf16_t*       h_out = (t & 1) ? h0g : h1g;

    auto issue_tiles = [&](int buf, int k0) {
      #pragma unroll
      for (int rr = 0; rr < 128; rr += 64) {
        const int nl = r + rr;
        copy16_async(sAb + buf * 4096 + nl * 32 + cc, h_in + (long)nl * H_ + k0 + cc);
        const int gr = (nl >> 5) * H_ + nb * 32 + (nl & 31);
        copy16_async(sBb + buf * 4096 + nl * 32 + cc, whh + (long)gr * H_ + k0 + cc);
      }
    };

    f32x8 acc[2][4] = {};
    issue_tiles(0, 0);
    for (int k0 = 0, it = 0; k0 < H_; k0 += 32, ++it) {
      const int cur = it & 1;
      wait_async_all();       // my async fills for buf[cur] landed
      __syncthreads();        // everyone's landed; everyone done reading buf[1-cur]
      if (k0 + 32 < H_) issue_tiles(1 - cur, k0 + 32);
      const bf16_t* sA = sAb + cur * 4096;
      const bf16_t* sB = sBb + cur * 4096;
      bf16x16 af[2], bfv[4];
      #pragma unroll
      for (int i = 0; i < 2; ++i) af[i] = load_a_frag(sA, wm + i * 16, lane);
      #pragma unroll
      for (int j = 0; j < 4; ++j) bfv[j] = load_b_frag(sB, wn + j * 16, lane);
      #pragma unroll
      for (int i = 0; i < 2; ++i)
        #pragma unroll
        for (int j = 0; j < 4; ++j)
          acc[i][j] = __builtin_amdgcn_wmma_f32_16x16x32_bf16(
              false, af[i], false, bfv[j], (short)0, acc[i][j], false, false);
    }
    __syncthreads();   // all tile reads done before gates overwrite the aliased LDS

    #pragma unroll
    for (int i = 0; i < 2; ++i)
      #pragma unroll
      for (int j = 0; j < 4; ++j) {
        const int nc = wn + j * 16 + (lane & 15);
        const int mr = wm + i * 16 + (lane >> 4) * 8;
        #pragma unroll
        for (int rr = 0; rr < 8; ++rr) sg[(mr + rr) * 128 + nc] = acc[i][j][rr];
      }
    __syncthreads();

    for (int idx = tid; idx < 128 * 32; idx += 256) {
      const int b  = idx >> 5, hl = idx & 31;
      const int hg = nb * 32 + hl;
      const long xrow = ((long)b * S_ + t) * G_;
      float iv = sg[b * 128 +      hl] + (float)xp[xrow +        hg] + b_hh[       hg];
      float fv = sg[b * 128 + 32 + hl] + (float)xp[xrow +   H_ + hg] + b_hh[  H_ + hg];
      float gv = sg[b * 128 + 64 + hl] + (float)xp[xrow + 2*H_ + hg] + b_hh[2*H_ + hg];
      float ov = sg[b * 128 + 96 + hl] + (float)xp[xrow + 3*H_ + hg] + b_hh[3*H_ + hg];
      const float is = 1.f / (1.f + __expf(-iv));
      const float fs = 1.f / (1.f + __expf(-fv));
      const float gt = tanhf(gv);
      const float os = 1.f / (1.f + __expf(-ov));
      const float cold = c[b * H_ + hg];
      const float hold = (float)h_in[b * H_ + hg];
      const float cn = fs * cold + is * gt;
      const float hn = os * tanhf(cn);
      const float m  = (float)mask[b * S_ + t];
      h_out[b * H_ + hg] = (bf16_t)(m * hn + (1.f - m) * hold);
      c[b * H_ + hg]     = m * cn + (1.f - m) * cold;
    }

    // order this step's h/c writes before any block starts step t+1
    grid_barrier(bar, 16u * (unsigned)(t + 1), tid);
  }
}

// ---------------- Final FC: out = relu( h @ W_fc^T + b_fc ) ----------------
__global__ __launch_bounds__(256)
void fc_kernel(const bf16_t* __restrict__ h, const bf16_t* __restrict__ wfc,
               const float* __restrict__ b_fc, float* __restrict__ out)
{
  __shared__ __align__(16) bf16_t sA[128 * 32];
  __shared__ __align__(16) bf16_t sB[128 * 32];
  const int tid = threadIdx.x, lane = tid & 31, wave = tid >> 5;
  const int wm = (wave & 3) * 32, wn = (wave >> 2) * 64;

  f32x8 acc[2][4] = {};
  for (int k0 = 0; k0 < H_; k0 += 32) {
    load_tile_bf16(sA, h + k0, H_, tid);
    load_tile_bf16(sB, wfc + k0, H_, tid);
    __syncthreads();
    bf16x16 af[2], bfv[4];
    #pragma unroll
    for (int i = 0; i < 2; ++i) af[i] = load_a_frag(sA, wm + i * 16, lane);
    #pragma unroll
    for (int j = 0; j < 4; ++j) bfv[j] = load_b_frag(sB, wn + j * 16, lane);
    #pragma unroll
    for (int i = 0; i < 2; ++i)
      #pragma unroll
      for (int j = 0; j < 4; ++j)
        acc[i][j] = __builtin_amdgcn_wmma_f32_16x16x32_bf16(
            false, af[i], false, bfv[j], (short)0, acc[i][j], false, false);
    __syncthreads();
  }
  #pragma unroll
  for (int i = 0; i < 2; ++i)
    #pragma unroll
    for (int j = 0; j < 4; ++j) {
      const int nc = wn + j * 16 + (lane & 15);
      const int mr = wm + i * 16 + (lane >> 4) * 8;
      const float bias = b_fc[nc];
      #pragma unroll
      for (int rr = 0; rr < 8; ++rr)
        out[(mr + rr) * O_ + nc] = fmaxf(acc[i][j][rr] + bias, 0.f);
    }
}

// ---------------- small helpers ----------------
__global__ void cvt_kernel(const float* __restrict__ s, bf16_t* __restrict__ d, int n) {
  int i = blockIdx.x * 256 + threadIdx.x;
  if (i < n) d[i] = (bf16_t)s[i];
}
__global__ void zero_f32_kernel(float* __restrict__ p, int n) {
  int i = blockIdx.x * 256 + threadIdx.x;
  if (i < n) p[i] = 0.f;
}
__global__ void zero_bf16_kernel(bf16_t* __restrict__ p, int n) {
  int i = blockIdx.x * 256 + threadIdx.x;
  if (i < n) p[i] = (bf16_t)0.f;
}

// ---------------- host launcher ----------------
extern "C" void kernel_launch(void* const* d_in, const int* in_sizes, int n_in,
                              void* d_out, int out_size, void* d_ws, size_t ws_size,
                              hipStream_t stream) {
  const float* x    = (const float*)d_in[0];
  const int*   mask = (const int*)  d_in[1];
  const float* Wih  = (const float*)d_in[2];
  const float* Whh  = (const float*)d_in[3];
  const float* bih  = (const float*)d_in[4];
  const float* bhh  = (const float*)d_in[5];
  const float* Wfc  = (const float*)d_in[6];
  const float* bfc  = (const float*)d_in[7];
  float* out = (float*)d_out;

  char* ws = (char*)d_ws;
  size_t off = 0;
  auto alloc = [&](size_t bytes) -> char* {
    char* p = ws + off;
    off = (off + bytes + 255) & ~(size_t)255;
    return p;
  };
  bf16_t*  xp  = (bf16_t*)alloc((size_t)B_ * S_ * G_ * 2);   // 512 MB bf16 x_proj
  bf16_t*  whh = (bf16_t*)alloc((size_t)G_ * H_ * 2);        // 2 MB (L2-resident)
  bf16_t*  wih = (bf16_t*)alloc((size_t)G_ * I_ * 2);
  bf16_t*  wfc = (bf16_t*)alloc((size_t)O_ * H_ * 2);
  bf16_t*  hb  = (bf16_t*)alloc((size_t)2 * B_ * H_ * 2);    // h0 | h1 (bf16)
  float*   c   = (float*) alloc((size_t)B_ * H_ * 4);
  unsigned* bar = (unsigned*)alloc(256);
  bf16_t* h0 = hb, *h1 = hb + B_ * H_;

  zero_bf16_kernel<<<(2 * B_ * H_ + 255) / 256, 256, 0, stream>>>(hb, 2 * B_ * H_);
  zero_f32_kernel <<<(B_ * H_ + 255) / 256, 256, 0, stream>>>(c, B_ * H_);
  zero_f32_kernel <<<1, 256, 0, stream>>>((float*)bar, 64);   // reset barrier each call
  cvt_kernel<<<(G_ * I_ + 255) / 256, 256, 0, stream>>>(Wih, wih, G_ * I_);
  cvt_kernel<<<(G_ * H_ + 255) / 256, 256, 0, stream>>>(Whh, whh, G_ * H_);
  cvt_kernel<<<(O_ * H_ + 255) / 256, 256, 0, stream>>>(Wfc, wfc, O_ * H_);

  // Phase 1: fully parallel input projection (1024 x 16 blocks)
  dim3 g1(B_ * S_ / 128, G_ / 128);
  xproj_gemm_kernel<<<g1, 256, 0, stream>>>(x, wih, bih, xp);

  // All 1024 recurrent steps in one persistent launch (16 co-resident blocks)
  lstm_persistent_kernel<<<16, 256, 0, stream>>>(h0, h1, c, whh, xp, bhh, mask, bar);

  // After t=1023 (odd), the last write went to h0.
  fc_kernel<<<1, 256, 0, stream>>>(h0, wfc, bfc, out);
}
